// GaWFRNNConv_61538291417378
// MI455X (gfx1250) — compile-verified, matrix operand order checked by reading, not run on hardware
//
#include <hip/hip_runtime.h>
#include <hip/hip_bf16.h>

typedef __attribute__((ext_vector_type(16))) _Float16 v16h;
typedef __attribute__((ext_vector_type(8)))  _Float16 v8h;
typedef __attribute__((ext_vector_type(8)))  float    v8f;

#define BTN   64        // B*T
#define FDM   98        // feedback dim
#define HID   256
#define INP   9216
#define COMB  9472
#define KPAD  128       // FD padded to multiple of 32 for f16 WMMA K
#define NCOLG 74        // 9472 / 128
#define NCC   62
#define NPP   36

// ---------------------------------------------------------------------------
// CDNA5 async global->LDS copy (16B per lane, tracked on ASYNCcnt).
// VDST = per-lane LDS byte offset (low 32 bits of generic LDS pointer),
// VADDR = 64-bit global address, saddr = off  (GV mode, §15.18.3).
// ---------------------------------------------------------------------------
__device__ __forceinline__ void async_cp16(unsigned lds_off, const void* gaddr) {
  asm volatile("global_load_async_to_lds_b128 %0, %1, off"
               :: "v"(lds_off), "v"((unsigned long long)(uintptr_t)gaddr)
               : "memory");
}
__device__ __forceinline__ void async_wait0() {
  asm volatile("s_wait_asynccnt 0" ::: "memory");
}

// fast sigmoid: v_exp_f32 + v_rcp_f32 (no IEEE divide expansion)
__device__ __forceinline__ float fast_sigmoid(float x) {
  return __builtin_amdgcn_rcpf(1.f + __expf(-x));
}

// ---------------------------------------------------------------------------
// Kernel A: conv1 (2->32, 3x3 SAME) + maxpool2 + LayerNorm(CHW) + ReLU
// one block per image; input image async-staged into LDS
// ---------------------------------------------------------------------------
__global__ __launch_bounds__(1024)
void k_conv1(const float* __restrict__ x, const float* __restrict__ w1,
             const float* __restrict__ b1, const float* __restrict__ ln1s,
             const float* __restrict__ ln1b, float* __restrict__ act1) {
  extern __shared__ float simg[];               // 2*96*96 = 18432 floats
  __shared__ float rs[1024], rq[1024];
  const int bt = blockIdx.x, tid = threadIdx.x;
  const char* img = (const char*)(x + (size_t)bt * 18432);
  const unsigned lbase = (unsigned)(uintptr_t)(void*)simg;
  for (int off = tid * 16; off < 18432 * 4; off += 1024 * 16)
    async_cp16(lbase + off, img + off);
  async_wait0();
  __syncthreads();
  float lsum = 0.f, lsq = 0.f;
  float* out = act1 + (size_t)bt * 73728;
  for (int idx = tid; idx < 73728; idx += 1024) {
    const int oc = idx / 2304, rem = idx % 2304, py = rem / 48, px = rem % 48;
    const float* w = w1 + oc * 18;
    const float bias = b1[oc];
    float mx = -3.4e38f;
    for (int dy = 0; dy < 2; ++dy)
      for (int dx = 0; dx < 2; ++dx) {
        const int y = 2 * py + dy, x0 = 2 * px + dx;
        float c = bias;
        for (int ic = 0; ic < 2; ++ic)
          for (int ky = 0; ky < 3; ++ky) {
            const int yy = y + ky - 1;
            if (yy < 0 || yy >= 96) continue;
            for (int kx = 0; kx < 3; ++kx) {
              const int xx = x0 + kx - 1;
              if (xx < 0 || xx >= 96) continue;
              c += simg[ic * 9216 + yy * 96 + xx] * w[(ic * 3 + ky) * 3 + kx];
            }
          }
        mx = fmaxf(mx, c);
      }
    out[idx] = mx; lsum += mx; lsq += mx * mx;
  }
  rs[tid] = lsum; rq[tid] = lsq; __syncthreads();
  for (int k = 512; k > 0; k >>= 1) {
    if (tid < k) { rs[tid] += rs[tid + k]; rq[tid] += rq[tid + k]; }
    __syncthreads();
  }
  const float mean = rs[0] * (1.f / 73728.f);
  const float var  = rq[0] * (1.f / 73728.f) - mean * mean;
  const float inv  = rsqrtf(var + 1e-5f);
  for (int idx = tid; idx < 73728; idx += 1024) {
    const float v = (out[idx] - mean) * inv * ln1s[idx] + ln1b[idx];
    out[idx] = fmaxf(v, 0.f);
  }
}

// ---------------------------------------------------------------------------
// Kernel B: conv2 (32->64, 3x3 SAME) + maxpool4 + LayerNorm(CHW) + ReLU
// async-stages the full 32x48x48 activation (288 KB) in CDNA5's 320 KB LDS
// ---------------------------------------------------------------------------
__global__ __launch_bounds__(1024)
void k_conv2(const float* __restrict__ act1, const float* __restrict__ w2,
             const float* __restrict__ b2, const float* __restrict__ ln2s,
             const float* __restrict__ ln2b, float* __restrict__ seq) {
  extern __shared__ float a1[];                 // 32*48*48 = 73728 floats
  __shared__ float rs[1024], rq[1024];
  const int bt = blockIdx.x, tid = threadIdx.x;
  const char* src = (const char*)(act1 + (size_t)bt * 73728);
  const unsigned lbase = (unsigned)(uintptr_t)(void*)a1;
  for (int off = tid * 16; off < 73728 * 4; off += 1024 * 16)   // 18 iters
    async_cp16(lbase + off, src + off);
  async_wait0();
  __syncthreads();
  float lsum = 0.f, lsq = 0.f;
  float* out = seq + (size_t)bt * INP;
  for (int idx = tid; idx < INP; idx += 1024) {
    const int oc = idx / 144, rem = idx % 144, py = rem / 12, px = rem % 12;
    const float* w = w2 + oc * 288;
    const float bias = b2[oc];
    float mx = -3.4e38f;
    for (int dy = 0; dy < 4; ++dy)
      for (int dx = 0; dx < 4; ++dx) {
        const int y = 4 * py + dy, x0 = 4 * px + dx;
        float c = bias;
        for (int ic = 0; ic < 32; ++ic) {
          const float* ap = a1 + ic * 2304;
          const float* wp = w + ic * 9;
          for (int ky = 0; ky < 3; ++ky) {
            const int yy = y + ky - 1;
            if (yy < 0 || yy >= 48) continue;
            for (int kx = 0; kx < 3; ++kx) {
              const int xx = x0 + kx - 1;
              if (xx < 0 || xx >= 48) continue;
              c += ap[yy * 48 + xx] * wp[ky * 3 + kx];
            }
          }
        }
        mx = fmaxf(mx, c);
      }
    out[idx] = mx; lsum += mx; lsq += mx * mx;
  }
  rs[tid] = lsum; rq[tid] = lsq; __syncthreads();
  for (int k = 512; k > 0; k >>= 1) {
    if (tid < k) { rs[tid] += rs[tid + k]; rq[tid] += rq[tid + k]; }
    __syncthreads();
  }
  const float mean = rs[0] * (1.f / 9216.f);
  const float var  = rq[0] * (1.f / 9216.f) - mean * mean;
  const float inv  = rsqrtf(var + 1e-5f);
  for (int idx = tid; idx < INP; idx += 1024) {
    const float v = (out[idx] - mean) * inv * ln2s[idx] + ln2b[idx];
    out[idx] = fmaxf(v, 0.f);
  }
}

// ---------------------------------------------------------------------------
// Prep kernels: V -> f16, A[bt] = f16(U * fb[bt]) padded to K=128, ih = b_ih
// ---------------------------------------------------------------------------
__global__ void k_prep_v(const float* __restrict__ V, _Float16* __restrict__ Vh) {
  const int i = blockIdx.x * 256 + threadIdx.x;
  if (i < FDM * COMB) Vh[i] = (_Float16)V[i];
}

__global__ void k_prep_a(const float* __restrict__ U, const float* __restrict__ fb,
                         _Float16* __restrict__ Af) {
  const int i = blockIdx.x * 256 + threadIdx.x;
  if (i >= BTN * HID * KPAD) return;
  const int bt = i / (HID * KPAD);
  const int r  = i % (HID * KPAD);
  const int h = r / KPAD, f = r % KPAD;
  float v = 0.f;
  if (f < FDM) v = U[h * FDM + f] * fb[bt * FDM + f];
  Af[i] = (_Float16)v;
}

__global__ void k_prep_ih(const float* __restrict__ bih, float* __restrict__ ih) {
  const int i = blockIdx.x * 256 + threadIdx.x;
  if (i < BTN * HID) ih[i] = bih[i & (HID - 1)];
}

// ---------------------------------------------------------------------------
// Kernel C: gate GEMM  Gpre = (U*fb) @ V  via v_wmma_f32_16x16x32_f16.
//   cols <  9216 : fuse sigmoid * W_ih * x  -> row-reduce into ih[bt,h]
//   cols >= 9216 : store sigmoid * W_hh     -> Ghw[bt,h,j] for the scan
// grid = (64 bt, 74 col-groups of 128); block = 256 (8 waves, 16 cols each)
// ---------------------------------------------------------------------------
__global__ __launch_bounds__(256)
void k_gate(const _Float16* __restrict__ Af, const _Float16* __restrict__ Vh,
            const float* __restrict__ seq, const float* __restrict__ Wih,
            const float* __restrict__ Whh, float* __restrict__ ih,
            float* __restrict__ Ghw) {
  __shared__ __align__(16) _Float16 Ah[HID * KPAD];   // 64 KB
  __shared__ __align__(16) _Float16 Bt[128 * KPAD];   // 32 KB (B transposed)
  __shared__ float red[HID];
  const int bt = blockIdx.x, cg = blockIdx.y;
  const int tid = threadIdx.x;
  const int c0 = cg * 128;
  const bool ihpath = (cg < 72);

  // async-stage A tile (contiguous 64 KB; f=98..127 pre-zeroed in prep)
  {
    const char* gsrc = (const char*)(Af + (size_t)bt * HID * KPAD);
    const unsigned lbase = (unsigned)(uintptr_t)(void*)Ah;
    for (int off = tid * 16; off < HID * KPAD * 2; off += 256 * 16)  // 16 iters
      async_cp16(lbase + off, gsrc + off);
  }
  // stage B tile transposed while the async copy is in flight
  for (int i = tid; i < 128 * KPAD; i += 256) {
    const int k = i >> 7, c = i & 127;
    _Float16 v = (_Float16)0.f;
    if (k < FDM) v = Vh[(size_t)k * COMB + c0 + c];
    Bt[c * KPAD + k] = v;
  }
  for (int i = tid; i < HID; i += 256) red[i] = 0.f;
  async_wait0();
  __syncthreads();

  const int wave = tid >> 5, lane = tid & 31;
  const int ln = lane & 15, hs = lane >> 4;     // 16x16 fragment halves
  const int lcol = wave * 16 + ln;              // local column 0..127
  const int gcol = c0 + lcol;                   // global column in COMB
  const float xv = ihpath ? seq[(size_t)bt * INP + gcol] : 0.f;

  for (int ht = 0; ht < 16; ++ht) {             // 16 row-tiles of HID
    const int h0 = ht * 16;
    v8f acc = {};
#pragma unroll
    for (int k0 = 0; k0 < KPAD; k0 += 32) {
      const int kb = k0 + hs * 8;
      // A 16x32 f16 fragment: lane<16 -> K {0..7,16..23}, lane>=16 -> +8
      const v8h alo = *(const v8h*)&Ah[(h0 + ln) * KPAD + kb];
      const v8h ahi = *(const v8h*)&Ah[(h0 + ln) * KPAD + kb + 16];
      // B 32x16 f16 fragment (column-contiguous in Bt)
      const v8h blo = *(const v8h*)&Bt[lcol * KPAD + kb];
      const v8h bhi = *(const v8h*)&Bt[lcol * KPAD + kb + 16];
      v16h a, b;
#pragma unroll
      for (int j = 0; j < 8; ++j) {
        a[j] = alo[j]; a[j + 8] = ahi[j];
        b[j] = blo[j]; b[j + 8] = bhi[j];
      }
      acc = __builtin_amdgcn_wmma_f32_16x16x32_f16(
          false, a, false, b, (short)0, acc, false, false);
    }
    if (ihpath) {
#pragma unroll
      for (int r = 0; r < 8; ++r) {
        const int m = h0 + r + 8 * hs;          // D row per §7.12.2 layout
        const float g = fast_sigmoid(acc[r]);
        float val = g * Wih[(size_t)m * INP + gcol] * xv;
        // sum 16 columns held by the 16-lane half
        for (int off = 8; off > 0; off >>= 1) val += __shfl_xor(val, off, 16);
        if (ln == 0) atomicAdd(&red[m], val);
      }
    } else {
      const int j = gcol - INP;
#pragma unroll
      for (int r = 0; r < 8; ++r) {
        const int m = h0 + r + 8 * hs;
        const float g = fast_sigmoid(acc[r]);
        Ghw[((size_t)bt * HID + m) * HID + j] = g * Whh[(size_t)m * HID + j];
      }
    }
  }
  if (ihpath) {
    __syncthreads();
    if (tid < HID) atomicAdd(&ih[(size_t)bt * HID + tid], red[tid]);
  }
}

// ---------------------------------------------------------------------------
// Kernel E: sequential scan (only hh depends on h_{t-1}); 4 blocks (one per b)
// ---------------------------------------------------------------------------
__global__ __launch_bounds__(256)
void k_scan(const float* __restrict__ ih, const float* __restrict__ Ghw,
            const float* __restrict__ bhh, const float* __restrict__ lns,
            const float* __restrict__ lnb, float* __restrict__ mid) {
  __shared__ float hprev[HID];
  __shared__ float s1[HID], s2[HID];
  const int b = blockIdx.x, tid = threadIdx.x;
  hprev[tid] = 0.f;
  __syncthreads();
  for (int t = 0; t < 16; ++t) {
    const int bt = b * 16 + t;
    const float* grow = Ghw + ((size_t)bt * HID + tid) * HID;
    __builtin_prefetch(grow + HID, 0, 0);       // row of thread tid+1, same step
    float hh = bhh[tid];
#pragma unroll 4
    for (int j = 0; j < HID; ++j) hh += grow[j] * hprev[j];
    const float htv = tanhf(ih[(size_t)bt * HID + tid] + hh);
    __syncthreads();
    s1[tid] = htv; s2[tid] = htv * htv;
    __syncthreads();
    for (int k = 128; k > 0; k >>= 1) {
      if (tid < k) { s1[tid] += s1[tid + k]; s2[tid] += s2[tid + k]; }
      __syncthreads();
    }
    const float mean = s1[0] * (1.f / HID);
    const float var  = s2[0] * (1.f / HID) - mean * mean;
    float o = (htv - mean) * rsqrtf(var + 1e-5f) * lns[tid] + lnb[tid];
    o = fmaxf(o, 0.f);
    mid[(size_t)bt * HID + tid] = o;
    __syncthreads();
    hprev[tid] = o;
    __syncthreads();
  }
}

// ---------------------------------------------------------------------------
// Kernel F: classifier heads; d_out = [char(4,16,62) | pos(4,16,36)] flat
// ---------------------------------------------------------------------------
__global__ __launch_bounds__(128)
void k_heads(const float* __restrict__ mid, const float* __restrict__ fccw,
             const float* __restrict__ fccb, const float* __restrict__ fcpw,
             const float* __restrict__ fcpb, float* __restrict__ out) {
  __shared__ float m[HID];
  const int bt = blockIdx.x, tid = threadIdx.x;
  m[tid] = mid[(size_t)bt * HID + tid];
  m[tid + 128] = mid[(size_t)bt * HID + tid + 128];
  __syncthreads();
  if (tid < NCC) {
    float a = fccb[tid];
    const float* w = fccw + tid * HID;
    for (int j = 0; j < HID; ++j) a += w[j] * m[j];
    out[bt * NCC + tid] = a;
  } else if (tid < NCC + NPP) {
    const int p = tid - NCC;
    float a = fcpb[p];
    const float* w = fcpw + p * HID;
    for (int j = 0; j < HID; ++j) a += w[j] * m[j];
    out[BTN * NCC + bt * NPP + p] = a;
  }
}

// ---------------------------------------------------------------------------
extern "C" void kernel_launch(void* const* d_in, const int* in_sizes, int n_in,
                              void* d_out, int out_size, void* d_ws, size_t ws_size,
                              hipStream_t stream) {
  const float* x    = (const float*)d_in[0];
  const float* fb   = (const float*)d_in[1];
  const float* c1w  = (const float*)d_in[2];
  const float* c1b  = (const float*)d_in[3];
  const float* ln1s = (const float*)d_in[4];
  const float* ln1b = (const float*)d_in[5];
  const float* c2w  = (const float*)d_in[6];
  const float* c2b  = (const float*)d_in[7];
  const float* ln2s = (const float*)d_in[8];
  const float* ln2b = (const float*)d_in[9];
  const float* Wih  = (const float*)d_in[10];
  const float* Whh  = (const float*)d_in[11];
  const float* bih  = (const float*)d_in[12];
  const float* bhh  = (const float*)d_in[13];
  const float* U    = (const float*)d_in[14];
  const float* V    = (const float*)d_in[15];
  const float* lnrs = (const float*)d_in[16];
  const float* lnrb = (const float*)d_in[17];
  const float* fccw = (const float*)d_in[18];
  const float* fccb = (const float*)d_in[19];
  const float* fcpw = (const float*)d_in[20];
  const float* fcpb = (const float*)d_in[21];

  char* ws = (char*)d_ws;
  size_t off = 0;
  float*    act1 = (float*)(ws + off);    off += (size_t)BTN * 73728 * 4;
  float*    seq  = (float*)(ws + off);    off += (size_t)BTN * INP * 4;
  _Float16* Vh   = (_Float16*)(ws + off); off += (size_t)FDM * COMB * 2;
  off = (off + 255) & ~(size_t)255;
  _Float16* Af   = (_Float16*)(ws + off); off += (size_t)BTN * HID * KPAD * 2;
  float*    ihb  = (float*)(ws + off);    off += (size_t)BTN * HID * 4;
  float*    Ghw  = (float*)(ws + off);    off += (size_t)BTN * HID * HID * 4;
  float*    mid  = (float*)(ws + off);    off += (size_t)BTN * HID * 4;

  k_conv1<<<BTN, 1024, 18432 * sizeof(float), stream>>>(x, c1w, c1b, ln1s, ln1b, act1);
  k_conv2<<<BTN, 1024, 73728 * sizeof(float), stream>>>(act1, c2w, c2b, ln2s, ln2b, seq);
  k_prep_v<<<(FDM * COMB + 255) / 256, 256, 0, stream>>>(V, Vh);
  k_prep_a<<<(BTN * HID * KPAD + 255) / 256, 256, 0, stream>>>(U, fb, Af);
  k_prep_ih<<<(BTN * HID + 255) / 256, 256, 0, stream>>>(bih, ihb);
  k_gate<<<dim3(BTN, NCOLG), 256, 0, stream>>>(Af, Vh, seq, Wih, Whh, ihb, Ghw);
  k_scan<<<4, 256, 0, stream>>>(ihb, Ghw, bhh, lnrs, lnrb, mid);
  k_heads<<<BTN, 128, 0, stream>>>(mid, fccw, fccb, fcpw, fcpb, (float*)d_out);
}
